// AnomalyTransformer_26611617366120
// MI455X (gfx1250) — matrix operand order repaired and checked by reading, not compiled, not run on hardware
//
#include <hip/hip_runtime.h>
#include <hip/hip_bf16.h>
#include <math.h>

typedef __attribute__((ext_vector_type(16))) _Float16 v16h;
typedef __attribute__((ext_vector_type(8)))  _Float16 v8h;
typedef __attribute__((ext_vector_type(8)))  float    v8f;

#define N_SEQ  1024
#define DMODEL 512
#define NLAYER 3
#define BATCH  8

// ---------------------------------------------------------------------------
// f32 -> f16 conversion (elementwise)
// ---------------------------------------------------------------------------
__global__ void cvt_f16_kernel(const float* __restrict__ in,
                               _Float16* __restrict__ out, int n) {
  int i = blockIdx.x * blockDim.x + threadIdx.x;
  if (i < n) out[i] = (_Float16)in[i];
}

// ---------------------------------------------------------------------------
// f32 -> f16 transpose: out[n*dim + k] = (f16) in[k*dim + n]
// LDS-tiled (32x32), block (32,8).
// ---------------------------------------------------------------------------
__global__ void transpose_f16_kernel(const float* __restrict__ in,
                                     _Float16* __restrict__ out, int dim) {
  __shared__ _Float16 tile[32][33];
  const int bx = blockIdx.x * 32;   // column range of input
  const int by = blockIdx.y * 32;   // row range of input
  const int tx = threadIdx.x;
  const int ty = threadIdx.y;
#pragma unroll
  for (int i = 0; i < 32; i += 8)
    tile[ty + i][tx] = (_Float16)in[(long)(by + ty + i) * dim + bx + tx];
  __syncthreads();
#pragma unroll
  for (int i = 0; i < 32; i += 8)
    out[(long)(bx + ty + i) * dim + by + tx] = tile[tx][ty + i];
}

// ---------------------------------------------------------------------------
// WMMA GEMM (A row-major [M,K] f16) x (Bt row-major [N,K] f16, i.e. B^T):
//   C = scale * (A @ B) [+ bias] [ReLU]
// Every operand load is a contiguous 128-bit chunk.
// Each wave computes a 32x64 tile = 2x4 WMMA tiles (8 independent chains):
// 12 b128 loads per 8 WMMAs (1.5 loads/WMMA).
// TRANSC=1 (f16 only): store C transposed, C^T[n*ldc + m], as one v8h/acc.
// Batched via gridDim.z with element strides sA/sB/sC.
// ---------------------------------------------------------------------------
template<int OUT_F16, int RELU, int TRANSC>
__global__ void wmma_gemm_tn_kernel(const _Float16* __restrict__ A, int lda, long sA,
                                    const _Float16* __restrict__ Bt, int ldb, long sB,
                                    float* __restrict__ Cf, _Float16* __restrict__ Ch,
                                    int ldc, long sC,
                                    const float* __restrict__ bias, float scale, int K)
{
  const int wave = threadIdx.x >> 5;
  const int lane = threadIdx.x & 31;
  const int half = lane >> 4;        // 16-lane half of the wave
  const int l16  = lane & 15;
  const int tileN = blockIdx.x * 64;
  const int tileM = (blockIdx.y * 4 + wave) * 32;
  const int bb = blockIdx.z;
  A  += (long)bb * sA;
  Bt += (long)bb * sB;

  v8f acc[2][4];
#pragma unroll
  for (int mi = 0; mi < 2; ++mi)
#pragma unroll
    for (int ni = 0; ni < 4; ++ni) acc[mi][ni] = (v8f){};

  // A 16x32 layout: lane l16 = row, elements {k+half*8+0..7, k+16+half*8+0..7}
  const _Float16* aP0 = A + (long)(tileM + l16) * lda + half * 8;
  const _Float16* aP1 = aP0 + (long)16 * lda;
  // B^T layout: lane l16 = column, elements {k+half*8+0..7, k+16+half*8+0..7}
  const _Float16* bP[4];
#pragma unroll
  for (int ni = 0; ni < 4; ++ni)
    bP[ni] = Bt + (long)(tileN + ni * 16 + l16) * ldb + half * 8;

  for (int k0 = 0; k0 < K; k0 += 32) {
    union { v16h v; v8h h[2]; } a0, a1, b[4];
    a0.h[0] = *(const v8h*)(aP0);
    a0.h[1] = *(const v8h*)(aP0 + 16);
    a1.h[0] = *(const v8h*)(aP1);
    a1.h[1] = *(const v8h*)(aP1 + 16);
#pragma unroll
    for (int ni = 0; ni < 4; ++ni) {
      b[ni].h[0] = *(const v8h*)(bP[ni]);
      b[ni].h[1] = *(const v8h*)(bP[ni] + 16);
      bP[ni] += 32;
    }
    aP0 += 32; aP1 += 32;

#pragma unroll
    for (int ni = 0; ni < 4; ++ni) {
      acc[0][ni] = __builtin_amdgcn_wmma_f32_16x16x32_f16(false, a0.v, false, b[ni].v,
                                                          (short)0, acc[0][ni], false, false);
      acc[1][ni] = __builtin_amdgcn_wmma_f32_16x16x32_f16(false, a1.v, false, b[ni].v,
                                                          (short)0, acc[1][ni], false, false);
    }
  }

  // C layout per tile: VGPR j -> M = 8*half + j, N = l16
#pragma unroll
  for (int mi = 0; mi < 2; ++mi) {
#pragma unroll
    for (int ni = 0; ni < 4; ++ni) {
      const v8f& a = acc[mi][ni];
      const int ncol = tileN + ni * 16 + l16;
      const int m0   = tileM + mi * 16 + half * 8;
      const float badd = (bias != nullptr) ? bias[ncol] : 0.0f;
      if (TRANSC) {
        v8h outv;
#pragma unroll
        for (int j = 0; j < 8; ++j) {
          float v = a[j] * scale + badd;
          if (RELU) v = v > 0.0f ? v : 0.0f;
          outv[j] = (_Float16)v;
        }
        *(v8h*)(Ch + (long)bb * sC + (long)ncol * ldc + m0) = outv;
      } else {
#pragma unroll
        for (int j = 0; j < 8; ++j) {
          float v = a[j] * scale + badd;
          if (RELU) v = v > 0.0f ? v : 0.0f;
          const long off = (long)bb * sC + (long)(m0 + j) * ldc + ncol;
          if (OUT_F16) Ch[off] = (_Float16)v;
          else         Cf[off] = v;
        }
      }
    }
  }
}

// ---------------------------------------------------------------------------
// Softmax over the BATCH axis (faithful to reference: softmax(scores, axis=0))
// ---------------------------------------------------------------------------
__global__ void softmax_b_kernel(const float* __restrict__ s,
                                 _Float16* __restrict__ out, int nn) {
  int i = blockIdx.x * blockDim.x + threadIdx.x;
  if (i >= nn) return;
  float v[BATCH];
  float mx = -INFINITY;
#pragma unroll
  for (int b = 0; b < BATCH; ++b) {
    v[b] = s[(long)b * nn + i];
    mx = fmaxf(mx, v[b]);
  }
  float sum = 0.0f;
#pragma unroll
  for (int b = 0; b < BATCH; ++b) { v[b] = __expf(v[b] - mx); sum += v[b]; }
  const float inv = 1.0f / sum;
#pragma unroll
  for (int b = 0; b < BATCH; ++b)
    out[(long)b * nn + i] = (_Float16)(v[b] * inv);
}

// ---------------------------------------------------------------------------
// y = LayerNorm(a + r) * g + beta ; writes f32 and (if non-null) f16 copy.
// One block (128 threads) per row of D=512.
// ---------------------------------------------------------------------------
__global__ void add_ln_kernel(const float* __restrict__ a, const float* __restrict__ r,
                              const float* __restrict__ g, const float* __restrict__ beta,
                              float* __restrict__ outF, _Float16* __restrict__ outH)
{
  __shared__ float red0[4];
  __shared__ float red1[4];
  const int row = blockIdx.x;
  const int t = threadIdx.x;
  const float* ap = a + (long)row * DMODEL;
  const float* rp = r + (long)row * DMODEL;

  float x[4];
  float s = 0.0f, sq = 0.0f;
#pragma unroll
  for (int i = 0; i < 4; ++i) {
    const float v = ap[t + i * 128] + rp[t + i * 128];
    x[i] = v; s += v; sq += v * v;
  }
#pragma unroll
  for (int off = 16; off > 0; off >>= 1) {
    s  += __shfl_down(s,  off, 32);
    sq += __shfl_down(sq, off, 32);
  }
  if ((t & 31) == 0) { red0[t >> 5] = s; red1[t >> 5] = sq; }
  __syncthreads();
  const float ts = red0[0] + red0[1] + red0[2] + red0[3];
  const float tq = red1[0] + red1[1] + red1[2] + red1[3];
  const float mean = ts * (1.0f / DMODEL);
  const float var  = tq * (1.0f / DMODEL) - mean * mean;
  const float inv  = rsqrtf(var + 1e-5f);
#pragma unroll
  for (int i = 0; i < 4; ++i) {
    const int c = t + i * 128;
    const float v = (x[i] - mean) * inv * g[c] + beta[c];
    outF[(long)row * DMODEL + c] = v;
    if (outH) outH[(long)row * DMODEL + c] = (_Float16)v;
  }
}

// ---------------------------------------------------------------------------
// out[row] = dot(h[row,:], Wc) + bc   (row = b*N + n), D = 512
// ---------------------------------------------------------------------------
__global__ void proj_kernel(const float* __restrict__ h, const float* __restrict__ Wc,
                            const float* __restrict__ bc, float* __restrict__ out)
{
  __shared__ float red[4];
  const int row = blockIdx.x;
  const int t = threadIdx.x;
  const float* hp = h + (long)row * DMODEL;
  float s = 0.0f;
#pragma unroll
  for (int i = 0; i < 4; ++i) { const int c = t + i * 128; s += hp[c] * Wc[c]; }
#pragma unroll
  for (int off = 16; off > 0; off >>= 1) s += __shfl_down(s, off, 32);
  if ((t & 31) == 0) red[t >> 5] = s;
  __syncthreads();
  if (t == 0) out[row] = red[0] + red[1] + red[2] + red[3] + bc[0];
}

// ---------------------------------------------------------------------------
extern "C" void kernel_launch(void* const* d_in, const int* in_sizes, int n_in,
                              void* d_out, int out_size, void* d_ws, size_t ws_size,
                              hipStream_t stream)
{
  (void)in_sizes; (void)n_in; (void)out_size; (void)ws_size;
  const float* x    = (const float*)d_in[0];
  const float* Wq   = (const float*)d_in[1];
  const float* Wk   = (const float*)d_in[2];
  const float* Wv   = (const float*)d_in[3];
  // d_in[4] = Ws: only feeds the prior P, which does not affect the output.
  const float* ln1g = (const float*)d_in[5];
  const float* ln1b = (const float*)d_in[6];
  const float* ffW  = (const float*)d_in[7];
  const float* ffb  = (const float*)d_in[8];
  const float* ln2g = (const float*)d_in[9];
  const float* ln2b = (const float*)d_in[10];
  const float* Wc   = (const float*)d_in[11];
  const float* bc   = (const float*)d_in[12];

  size_t o = 0;
  auto alloc = [&](size_t bytes) -> char* {
    char* p = (char*)d_ws + o;
    o += (bytes + 255) & ~(size_t)255;
    return p;
  };

  const long ND = (long)BATCH * N_SEQ * DMODEL;   // 4,194,304
  const long NN = (long)N_SEQ * N_SEQ;            // 1,048,576 (per batch)
  const long DD = (long)DMODEL * DMODEL;

  float*    hbuf   = (float*)   alloc(ND * 4);
  _Float16* h16    = (_Float16*)alloc(ND * 2);
  _Float16* q16    = (_Float16*)alloc(ND * 2);
  _Float16* k16    = (_Float16*)alloc(ND * 2);
  _Float16* v16t   = (_Float16*)alloc(ND * 2);            // V^T: [D, B*N]
  float*    scores = (float*)   alloc((long)BATCH * NN * 4);  // 32 MB, reused
  _Float16* s16    = (_Float16*)alloc((long)BATCH * NN * 2);
  float*    zbuf   = (float*)   alloc(ND * 4);
  _Float16* z16    = (_Float16*)alloc(ND * 2);
  _Float16* wq16t  = (_Float16*)alloc(DD * 2);            // W^T [out,in]
  _Float16* wk16t  = (_Float16*)alloc(DD * 2);
  _Float16* wv16t  = (_Float16*)alloc(DD * 2);
  _Float16* wf16t  = (_Float16*)alloc(DD * 2);
  float*    attnF  = scores;        // reuse scores[0 .. 16MB) once S16 is built
  float*    ffF    = scores + ND;   // reuse scores[16MB .. 32MB)

  // h = x (f32 residual path) and f16 copy for WMMA
  hipMemcpyAsync(hbuf, x, ND * sizeof(float), hipMemcpyDeviceToDevice, stream);
  cvt_f16_kernel<<<(int)((ND + 255) / 256), 256, 0, stream>>>(x, h16, (int)ND);

  const float inv_sqrt_d = 0.044194173824159216f;  // 1/sqrt(512)
  const dim3 blk(128, 1, 1);                       // 4 waves; each wave: 32x64 tile
  const dim3 gT(DMODEL / 32, DMODEL / 32, 1);      // weight transpose
  const dim3 bT(32, 8, 1);
  const dim3 gW(DMODEL / 64, (BATCH * N_SEQ) / 128, 1);    // (8, 64)
  const dim3 gS(N_SEQ / 64, N_SEQ / 128, BATCH);           // (16, 8, 8)
  const dim3 gA(DMODEL / 64, N_SEQ / 128, BATCH);          // (8, 8, 8)

  for (int l = 0; l < NLAYER; ++l) {
    const float* Wql = Wq  + (long)l * DD;
    const float* Wkl = Wk  + (long)l * DD;
    const float* Wvl = Wv  + (long)l * DD;
    const float* Wfl = ffW + (long)l * DD;
    transpose_f16_kernel<<<gT, bT, 0, stream>>>(Wql, wq16t, DMODEL);
    transpose_f16_kernel<<<gT, bT, 0, stream>>>(Wkl, wk16t, DMODEL);
    transpose_f16_kernel<<<gT, bT, 0, stream>>>(Wvl, wv16t, DMODEL);
    transpose_f16_kernel<<<gT, bT, 0, stream>>>(Wfl, wf16t, DMODEL);

    // Q, K = h @ W  ([B*N,512] x [512,512] -> f16, row-major)
    wmma_gemm_tn_kernel<1,0,0><<<gW, blk, 0, stream>>>(h16, DMODEL, 0,
        wq16t, DMODEL, 0, nullptr, q16, DMODEL, 0, nullptr, 1.0f, DMODEL);
    wmma_gemm_tn_kernel<1,0,0><<<gW, blk, 0, stream>>>(h16, DMODEL, 0,
        wk16t, DMODEL, 0, nullptr, k16, DMODEL, 0, nullptr, 1.0f, DMODEL);
    // V = h @ Wv, stored TRANSPOSED: V^T[d, b*N+m], ldc = B*N
    wmma_gemm_tn_kernel<1,0,1><<<gW, blk, 0, stream>>>(h16, DMODEL, 0,
        wv16t, DMODEL, 0, nullptr, v16t, BATCH * N_SEQ, 0, nullptr, 1.0f, DMODEL);

    // scores[b] = Q[b] @ K[b]^T / sqrt(D)   (f32)
    wmma_gemm_tn_kernel<0,0,0><<<gS, blk, 0, stream>>>(q16, DMODEL, (long)N_SEQ * DMODEL,
        k16, DMODEL, (long)N_SEQ * DMODEL,
        scores, nullptr, N_SEQ, NN, nullptr, inv_sqrt_d, DMODEL);

    // S = softmax over batch axis (dim 0), cast to f16
    softmax_b_kernel<<<(int)((NN + 255) / 256), 256, 0, stream>>>(scores, s16, (int)NN);

    // attn[b] = S[b] @ V[b] : B^T = V^T[d, b*N + m] with ldb = B*N, stride N
    wmma_gemm_tn_kernel<0,0,0><<<gA, blk, 0, stream>>>(s16, N_SEQ, NN,
        v16t, BATCH * N_SEQ, (long)N_SEQ,
        attnF, nullptr, DMODEL, (long)N_SEQ * DMODEL, nullptr, 1.0f, N_SEQ);

    // z = LN1(attn + h)
    add_ln_kernel<<<BATCH * N_SEQ, 128, 0, stream>>>(attnF, hbuf,
        ln1g + (long)l * DMODEL, ln1b + (long)l * DMODEL, zbuf, z16);

    // ff = relu(z @ ffW + ffb)
    wmma_gemm_tn_kernel<0,1,0><<<gW, blk, 0, stream>>>(z16, DMODEL, 0,
        wf16t, DMODEL, 0, ffF, nullptr, DMODEL, 0,
        ffb + (long)l * DMODEL, 1.0f, DMODEL);

    // h = LN2(ff + z)   (f32 + f16 for next layer)
    add_ln_kernel<<<BATCH * N_SEQ, 128, 0, stream>>>(ffF, zbuf,
        ln2g + (long)l * DMODEL, ln2b + (long)l * DMODEL, hbuf, h16);
  }

  // out[b,n] = h[b,n,:] . Wc + bc
  proj_kernel<<<BATCH * N_SEQ, 128, 0, stream>>>(hbuf, Wc, bc, (float*)d_out);
}
